// GenericTransformer_3994319586133
// MI455X (gfx1250) — matrix operand order repaired and checked by reading, not compiled, not run on hardware
//
#include <hip/hip_runtime.h>
#include <hip/hip_bf16.h>
#include <math.h>

// ---------------------------------------------------------------------------
// GPT-2 small forward for gfx1250 (MI455X), wave32 + WMMA f16 (fp32 accum).
// 128x128 block tile, 32x64 wave tile (8 WMMA / K-step / wave),
// double-buffered LDS, b128-aligned fragment layouts.
// ---------------------------------------------------------------------------

typedef __attribute__((ext_vector_type(16))) _Float16 v16h;
typedef __attribute__((ext_vector_type(8)))  _Float16 v8h;
typedef __attribute__((ext_vector_type(8)))  float    v8f;

#define BLK   256
#define TM    128
#define TN    128
#define TK    32
#define LDA_S 40   // halves; 80 B row stride: 16B-aligned chunks, conflict-free
#define LDB_T 40   // halves; B stored transposed [n][k]

// ---- WMMA fragment loads, per ISA 7.12.2 wave32 layouts --------------------
// A (16x32 f16): lane&15 = row M; lanes 0-15 hold K 0-7 / 16-23,
//                lanes 16-31 hold K 8-15 / 24-31.  Two aligned b128 loads.
__device__ __forceinline__ v16h frag_a(const _Float16* base, int lane) {
  const _Float16* r = base + (lane & 15) * LDA_S + (lane >> 4) * 8;
  const v8h lo = *(const v8h*)(r);
  const v8h hi = *(const v8h*)(r + 16);
  v16h a;
#pragma unroll
  for (int i = 0; i < 8; ++i) { a[i] = lo[i]; a[8 + i] = hi[i]; }
  return a;
}
// B (32x16 f16): lane&15 = col N; lanes 0-15 hold K 0-7 / 16-23, lanes 16-31 K 8-15 / 24-31.
// Bt is [n][k] so each lane reads two runs of 8 contiguous halves, 16 apart: two b128 loads.
__device__ __forceinline__ v16h frag_b(const _Float16* baseT, int lane) {
  const _Float16* r = baseT + (lane & 15) * LDB_T + (lane >> 4) * 8;
  const v8h lo = *(const v8h*)(r);
  const v8h hi = *(const v8h*)(r + 16);
  v16h b;
#pragma unroll
  for (int i = 0; i < 8; ++i) { b[i] = lo[i]; b[8 + i] = hi[i]; }
  return b;
}

// Cooperative staging of one (A 128x32, B 32x128) K-tile into LDS (fp32->f16).
__device__ __forceinline__ void stage_tiles(
    _Float16* __restrict__ As, _Float16* __restrict__ Bt,
    const float* __restrict__ A, const float* __restrict__ Bm,
    long long aoff, long long boff,
    int m0, int n0, int N, int lda, int ldb, int bstr, int k0,
    int arow, int acol, int brow, int bcol0)
{
  // A: 16 contiguous fp32 per thread -> 16 contiguous halves (b128 stores).
  const float* ap = A + aoff + (long long)(m0 + arow) * lda + (k0 + acol);
  _Float16* asp = As + arow * LDA_S + acol;
#pragma unroll
  for (int j = 0; j < 16; ++j) asp[j] = (_Float16)ap[j];

  // B: 16 fp32 per thread (same k, consecutive n), transposed b16 scatter.
  const long long brbase = boff + (long long)(k0 + brow) * ldb;
  if (bstr == 1 && n0 + TN <= N && (ldb & 3) == 0) {
    // Fast path: contiguous, aligned, fully in-range -> 4x float4 loads.
    const float* bp = Bm + brbase + n0 + bcol0;
    const float4 f0 = *(const float4*)(bp);
    const float4 f1 = *(const float4*)(bp + 4);
    const float4 f2 = *(const float4*)(bp + 8);
    const float4 f3 = *(const float4*)(bp + 12);
    const float t[16] = {f0.x, f0.y, f0.z, f0.w, f1.x, f1.y, f1.z, f1.w,
                         f2.x, f2.y, f2.z, f2.w, f3.x, f3.y, f3.z, f3.w};
#pragma unroll
    for (int j = 0; j < 16; ++j)
      Bt[(bcol0 + j) * LDB_T + brow] = (_Float16)t[j];
  } else {
#pragma unroll
    for (int j = 0; j < 16; ++j) {
      const int n = n0 + bcol0 + j;
      float bv = 0.0f;
      if (n < N) bv = Bm[brbase + (long long)n * bstr];
      Bt[(bcol0 + j) * LDB_T + brow] = (_Float16)bv;
    }
  }
}

// ---------------------------------------------------------------------------
// Generic batched GEMM: C[z] = epilogue(A[z](MxK) * B[z](KxN) + bias [+ res])
//   A element (m,k): A[aoff + m*lda + k]
//   B element (k,n): B[boff + k*ldb + n*bstr]
//   C element (m,n): C[coff + m*ldc + n]
//   batch z: zo = z / nInner, zi = z % nInner; off = zo*Outer + zi*Inner
// Requires M % 128 == 0, K % 32 == 0 (true at all call sites); N guarded.
// ---------------------------------------------------------------------------
__global__ void __launch_bounds__(BLK)
gemm_wmma_f16(const float* __restrict__ A, const float* __restrict__ Bm,
              const float* __restrict__ bias, const float* __restrict__ Res,
              float* __restrict__ Cm,
              int M, int N, int K,
              int lda, int ldb, int bstr, int ldc,
              long long aO, long long aI, long long bO, long long bI,
              long long cO, long long cI, int nInner, int fuseGelu)
{
  __shared__ _Float16 As[2][TM * LDA_S];   // 2 x 10 KB
  __shared__ _Float16 Bs[2][TN * LDB_T];   // 2 x 10 KB

  const int tid  = threadIdx.x;
  const int lane = tid & 31;
  const int wave = tid >> 5;
  const int wm   = wave & 3;   // 4 wave-rows of 32
  const int wn   = wave >> 2;  // 2 wave-cols of 64

  const int m0 = blockIdx.y * TM;
  const int n0 = blockIdx.x * TN;
  const int zo = blockIdx.z / nInner;
  const int zi = blockIdx.z % nInner;
  const long long aoff = (long long)zo * aO + (long long)zi * aI;
  const long long boff = (long long)zo * bO + (long long)zi * bI;
  const long long coff = (long long)zo * cO + (long long)zi * cI;

  const int arow  = tid >> 1, acol  = (tid & 1) * 16;  // 128 rows x 2 chunks
  const int brow  = tid >> 3, bcol0 = (tid & 7) * 16;  // 32 k-rows x 8 chunks

  v8f acc[2][4] = {};

  const int nk = K / TK;
  stage_tiles(As[0], Bs[0], A, Bm, aoff, boff, m0, n0, N, lda, ldb, bstr,
              0, arow, acol, brow, bcol0);
  __syncthreads();

  for (int kt = 0; kt < nk; ++kt) {
    const int p = kt & 1;
    if (kt + 1 < nk) {
      stage_tiles(As[p ^ 1], Bs[p ^ 1], A, Bm, aoff, boff, m0, n0, N,
                  lda, ldb, bstr, (kt + 1) * TK, arow, acol, brow, bcol0);
      __builtin_prefetch(A + aoff + (long long)(m0 + arow) * lda
                           + (kt + 2) * TK + acol, 0, 0);
    }

    const v16h a0 = frag_a(As[p] + (wm * 32) * LDA_S, lane);
    const v16h a1 = frag_a(As[p] + (wm * 32 + 16) * LDA_S, lane);
    const v16h b0 = frag_b(Bs[p] + (wn * 64) * LDB_T, lane);
    const v16h b1 = frag_b(Bs[p] + (wn * 64 + 16) * LDB_T, lane);
    const v16h b2 = frag_b(Bs[p] + (wn * 64 + 32) * LDB_T, lane);
    const v16h b3 = frag_b(Bs[p] + (wn * 64 + 48) * LDB_T, lane);

    acc[0][0] = __builtin_amdgcn_wmma_f32_16x16x32_f16(false, a0, false, b0,
                    (short)0, acc[0][0], false, false);
    acc[0][1] = __builtin_amdgcn_wmma_f32_16x16x32_f16(false, a0, false, b1,
                    (short)0, acc[0][1], false, false);
    acc[0][2] = __builtin_amdgcn_wmma_f32_16x16x32_f16(false, a0, false, b2,
                    (short)0, acc[0][2], false, false);
    acc[0][3] = __builtin_amdgcn_wmma_f32_16x16x32_f16(false, a0, false, b3,
                    (short)0, acc[0][3], false, false);
    acc[1][0] = __builtin_amdgcn_wmma_f32_16x16x32_f16(false, a1, false, b0,
                    (short)0, acc[1][0], false, false);
    acc[1][1] = __builtin_amdgcn_wmma_f32_16x16x32_f16(false, a1, false, b1,
                    (short)0, acc[1][1], false, false);
    acc[1][2] = __builtin_amdgcn_wmma_f32_16x16x32_f16(false, a1, false, b2,
                    (short)0, acc[1][2], false, false);
    acc[1][3] = __builtin_amdgcn_wmma_f32_16x16x32_f16(false, a1, false, b3,
                    (short)0, acc[1][3], false, false);

    __syncthreads();
  }

  // Epilogue: C/D layout — lane&15 = N col, lane>=16 => rows 8..15.
#pragma unroll
  for (int tm = 0; tm < 2; ++tm) {
#pragma unroll
    for (int tn = 0; tn < 4; ++tn) {
      const int n = n0 + wn * 64 + tn * 16 + (lane & 15);
      if (n >= N) continue;
      const int mb = m0 + wm * 32 + tm * 16 + ((lane >> 4) * 8);
      const float bv = bias ? bias[n] : 0.0f;
#pragma unroll
      for (int r = 0; r < 8; ++r) {
        float vv = acc[tm][tn][r] + bv;
        if (fuseGelu) vv = 0.5f * vv * (1.0f + erff(vv * 0.70710678118654752f));
        const long long idx = coff + (long long)(mb + r) * ldc + n;
        if (Res) vv += Res[idx];
        Cm[idx] = vv;
      }
    }
  }
}

// ---------------------------------------------------------------------------
// x[b,t,:] = wte[idx[b,t],:] + wpe[t,:]
// ---------------------------------------------------------------------------
__global__ void __launch_bounds__(BLK)
embed_k(const int* __restrict__ idx, const float* __restrict__ wte,
        const float* __restrict__ wpe, float* __restrict__ x)
{
  const int row = blockIdx.x;          // b*T + t
  const int t   = row & 1023;
  const int tok = idx[row];
  const float* pt = wte + (long long)tok * 768;
  const float* pe = wpe + (long long)t * 768;
  float* px = x + (long long)row * 768;
  for (int i = threadIdx.x; i < 768; i += BLK) px[i] = pt[i] + pe[i];
}

// ---------------------------------------------------------------------------
// Row LayerNorm over C=768 (wave32 shuffle + LDS block reduce).
// ---------------------------------------------------------------------------
__global__ void __launch_bounds__(BLK)
layernorm_k(const float* __restrict__ x, const float* __restrict__ w,
            const float* __restrict__ b, float* __restrict__ out)
{
  __shared__ float red[18];
  const int row = blockIdx.x;
  const float* px = x + (long long)row * 768;
  float s = 0.0f, s2 = 0.0f;
  for (int i = threadIdx.x; i < 768; i += BLK) {
    const float v = px[i]; s += v; s2 += v * v;
  }
#pragma unroll
  for (int off = 16; off > 0; off >>= 1) {
    s  += __shfl_down(s,  off, 32);
    s2 += __shfl_down(s2, off, 32);
  }
  const int wv = threadIdx.x >> 5, ln = threadIdx.x & 31;
  if (ln == 0) { red[wv] = s; red[8 + wv] = s2; }
  __syncthreads();
  if (threadIdx.x == 0) {
    float a = 0.0f, c = 0.0f;
#pragma unroll
    for (int i = 0; i < 8; ++i) { a += red[i]; c += red[8 + i]; }
    const float m = a * (1.0f / 768.0f);
    const float var = c * (1.0f / 768.0f) - m * m;
    red[16] = m; red[17] = rsqrtf(var + 1e-5f);
  }
  __syncthreads();
  const float m = red[16], inv = red[17];
  float* po = out + (long long)row * 768;
  for (int i = threadIdx.x; i < 768; i += BLK)
    po[i] = (px[i] - m) * inv * w[i] + b[i];
}

// ---------------------------------------------------------------------------
// In-place causal softmax on one score row of length T=1024 (scale = 1/8).
// blockIdx.x = (b*H + h)*T + q
// ---------------------------------------------------------------------------
__global__ void __launch_bounds__(BLK)
softmax_causal_k(float* __restrict__ att)
{
  __shared__ float red[9];
  const int row = blockIdx.x;
  const int q   = row & 1023;
  float* p = att + (long long)row * 1024;
  const float scale = 0.125f;  // 1/sqrt(64)
  const float NEG = -3.402823466e38f;

  float mx = NEG;
  for (int k = threadIdx.x; k < 1024; k += BLK) {
    const float v = (k <= q) ? p[k] * scale : NEG;
    mx = fmaxf(mx, v);
  }
#pragma unroll
  for (int off = 16; off > 0; off >>= 1) mx = fmaxf(mx, __shfl_down(mx, off, 32));
  const int wv = threadIdx.x >> 5, ln = threadIdx.x & 31;
  if (ln == 0) red[wv] = mx;
  __syncthreads();
  if (threadIdx.x == 0) {
    float m = red[0];
#pragma unroll
    for (int i = 1; i < 8; ++i) m = fmaxf(m, red[i]);
    red[8] = m;
  }
  __syncthreads();
  mx = red[8];

  float sum = 0.0f;
  for (int k = threadIdx.x; k < 1024; k += BLK) {
    const float v = (k <= q) ? expf(p[k] * scale - mx) : 0.0f;
    p[k] = v;
    sum += v;
  }
#pragma unroll
  for (int off = 16; off > 0; off >>= 1) sum += __shfl_down(sum, off, 32);
  if (ln == 0) red[wv] = sum;
  __syncthreads();
  if (threadIdx.x == 0) {
    float t = 0.0f;
#pragma unroll
    for (int i = 0; i < 8; ++i) t += red[i];
    red[8] = 1.0f / t;
  }
  __syncthreads();
  const float rs = red[8];
  for (int k = threadIdx.x; k < 1024; k += BLK) p[k] *= rs;
}

// ---------------------------------------------------------------------------
// Host side
// ---------------------------------------------------------------------------
static inline void launch_gemm(hipStream_t st, const float* A, const float* B,
                               const float* bias, const float* Res, float* C,
                               int M, int N, int K,
                               int lda, int ldb, int bstr, int ldc,
                               long long aO, long long aI, long long bO,
                               long long bI, long long cO, long long cI,
                               int nInner, int nBatch, int gelu)
{
  dim3 grid((N + TN - 1) / TN, M / TM, nBatch);
  gemm_wmma_f16<<<grid, BLK, 0, st>>>(A, B, bias, Res, C, M, N, K,
                                      lda, ldb, bstr, ldc,
                                      aO, aI, bO, bI, cO, cI, nInner, gelu);
}

extern "C" void kernel_launch(void* const* d_in, const int* in_sizes, int n_in,
                              void* d_out, int out_size, void* d_ws, size_t ws_size,
                              hipStream_t stream)
{
  (void)in_sizes; (void)n_in; (void)out_size; (void)ws_size;
  constexpr int B = 2, T = 1024, C = 768, H = 12, D = 64, L = 12;
  constexpr int V = 50257, F = 3072;
  constexpr int NT = B * T;
  constexpr long long NTC  = (long long)NT * C;        // 1.5M floats
  constexpr long long BHTT = (long long)B * H * T * T; // 25.2M floats

  const int*   idx  = (const int*)  d_in[0];
  const float* wte  = (const float*)d_in[1];
  const float* wpe  = (const float*)d_in[2];
  const float* ln1w = (const float*)d_in[3];
  const float* ln1b = (const float*)d_in[4];
  const float* Wq   = (const float*)d_in[5];
  const float* bq   = (const float*)d_in[6];
  const float* Wk   = (const float*)d_in[7];
  const float* bk   = (const float*)d_in[8];
  const float* Wv   = (const float*)d_in[9];
  const float* bv   = (const float*)d_in[10];
  const float* Wo   = (const float*)d_in[11];
  const float* bo   = (const float*)d_in[12];
  const float* ln2w = (const float*)d_in[13];
  const float* ln2b = (const float*)d_in[14];
  const float* Wfc  = (const float*)d_in[15];
  const float* bfc  = (const float*)d_in[16];
  const float* Wpr  = (const float*)d_in[17];
  const float* bpr  = (const float*)d_in[18];
  const float* lnfw = (const float*)d_in[19];
  const float* lnfb = (const float*)d_in[20];
  const float* Wlm  = (const float*)d_in[21];
  float* logits = (float*)d_out;

  // Workspace carve-out (fp32): x, h, q, k, v, y, att, fc  (~164 MB)
  float* x   = (float*)d_ws;
  float* h   = x  + NTC;
  float* q   = h  + NTC;
  float* kq  = q  + NTC;
  float* vq  = kq + NTC;
  float* y   = vq + NTC;
  float* att = y  + NTC;
  float* fc  = att + BHTT;

  embed_k<<<NT, BLK, 0, stream>>>(idx, wte, wpe, x);

  const long long sTC = (long long)T * C;   // per-batch activation stride
  const long long sTT = (long long)T * T;   // per-head score stride
  const long long sH  = (long long)H * sTT; // per-batch score stride

  for (int l = 0; l < L; ++l) {
    const float* pWq = Wq + (long long)l * C * C;
    const float* pWk = Wk + (long long)l * C * C;
    const float* pWv = Wv + (long long)l * C * C;
    const float* pWo = Wo + (long long)l * C * C;
    const float* pWf = Wfc + (long long)l * C * F;
    const float* pWp = Wpr + (long long)l * F * C;

    // h = LN1(x)
    layernorm_k<<<NT, BLK, 0, stream>>>(x, ln1w + l * C, ln1b + l * C, h);

    // q/k/v = h @ W + b   (M=2048, N=768, K=768)
    launch_gemm(stream, h, pWq, bq + l * C, nullptr, q,  NT, C, C,
                C, C, 1, C, 0, 0, 0, 0, 0, 0, 1, 1, 0);
    launch_gemm(stream, h, pWk, bk + l * C, nullptr, kq, NT, C, C,
                C, C, 1, C, 0, 0, 0, 0, 0, 0, 1, 1, 0);
    launch_gemm(stream, h, pWv, bv + l * C, nullptr, vq, NT, C, C,
                C, C, 1, C, 0, 0, 0, 0, 0, 0, 1, 1, 0);

    // scores[b,h] = q[b,:,h,:] @ k[b,:,h,:]^T   (M=T, N=T, K=D), batched b*H
    launch_gemm(stream, q, kq, nullptr, nullptr, att, T, T, D,
                C, 1, C, T, sTC, D, sTC, D, sH, sTT, H, B * H, 0);

    // causal softmax in place
    softmax_causal_k<<<B * H * T, BLK, 0, stream>>>(att);

    // y[b,:,h,:] = att[b,h] @ v[b,:,h,:]   (M=T, N=D, K=T), batched b*H
    launch_gemm(stream, att, vq, nullptr, nullptr, y, T, D, T,
                T, C, 1, C, sH, sTT, sTC, D, sTC, D, H, B * H, 0);

    // x = x + y @ Wo + bo
    launch_gemm(stream, y, pWo, bo + l * C, x, x, NT, C, C,
                C, C, 1, C, 0, 0, 0, 0, 0, 0, 1, 1, 0);

    // h = LN2(x)
    layernorm_k<<<NT, BLK, 0, stream>>>(x, ln2w + l * C, ln2b + l * C, h);

    // fc = gelu(h @ Wfc + bfc)   (N=3072)
    launch_gemm(stream, h, pWf, bfc + (long long)l * F, nullptr, fc, NT, F, C,
                C, F, 1, F, 0, 0, 0, 0, 0, 0, 1, 1, 1);

    // x = x + fc @ Wpr + bpr     (K=3072)
    launch_gemm(stream, fc, pWp, bpr + l * C, x, x, NT, C, F,
                F, C, 1, C, 0, 0, 0, 0, 0, 0, 1, 1, 0);
  }

  // h = LN_f(x); logits = h @ Wlm   (M=2048, N=50257, K=768)
  layernorm_k<<<NT, BLK, 0, stream>>>(x, lnfw, lnfb, h);
  launch_gemm(stream, h, Wlm, nullptr, nullptr, logits, NT, V, C,
              C, V, 1, V, 0, 0, 0, 0, 0, 0, 1, 1, 0);
}
